// InteractionBlock_48928267436051
// MI455X (gfx1250) — matrix operand order repaired and checked by reading, not compiled, not run on hardware
//
#include <hip/hip_runtime.h>

#define H128 128
#define NRr 6
#define NBb 8
#define SBFW 42

typedef __attribute__((ext_vector_type(16))) __bf16 v16bf;
typedef __attribute__((ext_vector_type(8)))  __bf16 v8bf;
typedef __attribute__((ext_vector_type(8)))  float  v8f;

__device__ __forceinline__ v16bf cat8(v8bf lo, v8bf hi) {
    return __builtin_shufflevector(lo, hi, 0,1,2,3,4,5,6,7,8,9,10,11,12,13,14,15);
}

__device__ __forceinline__ v8f wmma_bf16(v16bf a, v16bf b, v8f c) {
    return __builtin_amdgcn_wmma_f32_16x16x32_bf16(false, a, false, b, (short)0, c,
                                                   false, false);
}

__device__ __forceinline__ float silu_f(float x) {
    return x / (1.0f + __expf(-x));
}

__device__ __forceinline__ void zero8(v8f* a) {
#pragma unroll
    for (int t = 0; t < 8; ++t) {
        v8f z = {0.f, 0.f, 0.f, 0.f, 0.f, 0.f, 0.f, 0.f};
        a[t] = z;
    }
}

// ---- CDNA5 async global->LDS (ASYNCcnt) via inline asm ----
// Generic pointers into LDS carry the wave-relative LDS byte offset in
// bits[31:0] (aperture mapping, ISA 10.2), so truncation gives the VDST value.
__device__ __forceinline__ unsigned lds_off32(const void* p) {
    return (unsigned)(uintptr_t)p;
}
__device__ __forceinline__ void async_ld16(unsigned lds_off, const void* gsrc) {
    asm volatile("global_load_async_to_lds_b128 %0, %1, off"
                 :: "v"(lds_off), "v"(gsrc) : "memory");
}
__device__ __forceinline__ void async_wait0() {
    asm volatile("s_wait_asynccnt 0x0" ::: "memory");
}

// acc[8] += A(16x128, LDS bf16, row pitch `pitch` elems) @ Wt(global bf16 [128 n][128 k])
// A layout per ISA: lane holds row (lane&15); K halves split by lane>>4.
__device__ __forceinline__ void mm128(const __bf16* A, int pitch,
                                      const __bf16* __restrict__ Wt, v8f* acc) {
    const int lane = threadIdx.x & 31;
    const int g = lane >> 4, r = lane & 15;
#pragma unroll
    for (int ks = 0; ks < 4; ++ks) {
        const __bf16* ar = A + r * pitch + ks * 32;
        v16bf a = cat8(*(const v8bf*)(ar + 8 * g), *(const v8bf*)(ar + 16 + 8 * g));
#pragma unroll
        for (int t = 0; t < 8; ++t) {
            const __bf16* br = Wt + (t * 16 + r) * H128 + ks * 32 + 8 * g;
            v16bf b = cat8(*(const v8bf*)(br), *(const v8bf*)(br + 16));
            acc[t] = wmma_bf16(a, b, acc[t]);
        }
    }
}

// Store C-layout regs (M = v + 8*(lane>>4), N = t*16 + lane&15) into per-wave LDS
// A-scratch (16 rows x 136 pitch, bf16).
__device__ __forceinline__ void writeA(__bf16* Aw, const v8f* h) {
    const int lane = threadIdx.x & 31;
    const int g = lane >> 4, r = lane & 15;
#pragma unroll
    for (int t = 0; t < 8; ++t)
#pragma unroll
        for (int v = 0; v < 8; ++v)
            Aw[(v + 8 * g) * 136 + t * 16 + r] = (__bf16)h[t][v];
}

// out = silu(in @ Wt + b); in/out may alias.
__device__ __forceinline__ void linear_silu(__bf16* Aw, const v8f* in,
                                            const __bf16* __restrict__ Wt,
                                            const float* __restrict__ b, v8f* out) {
    writeA(Aw, in);
    zero8(out);
    mm128(Aw, 136, Wt, out);
    const int r = threadIdx.x & 15;
#pragma unroll
    for (int t = 0; t < 8; ++t)
#pragma unroll
        for (int v = 0; v < 8; ++v)
            out[t][v] = silu_f(out[t][v] + b[t * 16 + r]);
}

// ---------------- prep kernels ----------------

__global__ __launch_bounds__(256) void cvt_bf16_kernel(const float* __restrict__ src,
                                                       __bf16* __restrict__ dst, int n) {
    for (int i = blockIdx.x * 256 + threadIdx.x; i < n; i += gridDim.x * 256)
        dst[i] = (__bf16)src[i];
}

// src: [128 in][128 out] fp32 -> dst: [128 out][128 in] bf16
__global__ __launch_bounds__(256) void transpose_cvt_kernel(const float* __restrict__ src,
                                                            __bf16* __restrict__ dst) {
    int idx = blockIdx.x * 256 + threadIdx.x;  // 16384 total
    int n = idx >> 7, k = idx & 127;
    dst[n * H128 + k] = (__bf16)src[k * H128 + n];
}

__global__ __launch_bounds__(256) void zero_f32_kernel(float* __restrict__ p, long n) {
    for (long i = blockIdx.x * 256L + threadIdx.x; i < n; i += (long)gridDim.x * 256)
        p[i] = 0.f;
}

// ---------------- K1: edge in-projections ----------------
// x_ji = silu(x@Wji+bji); x_kj = silu(x@Wkj+bkj) * (rbf@lin_rbf_w)
__global__ __launch_bounds__(256) void k1_edge_in(
    const float* __restrict__ x, const float* __restrict__ rbf,
    const float* __restrict__ lin_rbf_w, const float* __restrict__ bkj,
    const float* __restrict__ bji, const __bf16* __restrict__ WtKJ,
    const __bf16* __restrict__ WtJI, float* __restrict__ x_ji,
    __bf16* __restrict__ x_kj, int E) {
    __shared__ __align__(16) __bf16 xA[128 * 136];
    __shared__ float rbf_s[128 * NRr];
    __shared__ float lrw_s[NRr * H128];
    __shared__ float bji_s[H128], bkj_s[H128];

    const int tid = threadIdx.x;
    const long base = (long)blockIdx.x * 128;

    {   // x tile -> bf16 LDS (2 threads per row)
        int row = tid >> 1, half = tid & 1;
        long gr = base + row;
        __bf16* dst = xA + row * 136 + half * 64;
        if (gr < E) {
            const float* src = x + gr * H128 + half * 64;
#pragma unroll
            for (int i = 0; i < 64; i += 4) {
                float4 v = *(const float4*)(src + i);
                dst[i + 0] = (__bf16)v.x; dst[i + 1] = (__bf16)v.y;
                dst[i + 2] = (__bf16)v.z; dst[i + 3] = (__bf16)v.w;
            }
        } else {
#pragma unroll
            for (int i = 0; i < 64; ++i) dst[i] = (__bf16)0.f;
        }
    }
    if (tid < H128) { bji_s[tid] = bji[tid]; bkj_s[tid] = bkj[tid]; }
    for (int i = tid; i < NRr * H128; i += 256) lrw_s[i] = lin_rbf_w[i];
    for (int i = tid; i < 128 * NRr; i += 256) {
        long gr = base + i / NRr;
        rbf_s[i] = (gr < E) ? rbf[base * NRr + i] : 0.f;
    }
    __syncthreads();

    const int wave = tid >> 5, lane = tid & 31, g = lane >> 4, r = lane & 15;
    const __bf16* Aw = xA + wave * 16 * 136;
    v8f acc[8];

    // ---- lin_ji ----
    zero8(acc);
    mm128(Aw, 136, WtJI, acc);
#pragma unroll
    for (int t = 0; t < 8; ++t)
#pragma unroll
        for (int v = 0; v < 8; ++v) {
            int col = t * 16 + r;
            int mrow = wave * 16 + v + 8 * g;
            long gr = base + mrow;
            float val = silu_f(acc[t][v] + bji_s[col]);
            if (gr < E) x_ji[gr * H128 + col] = val;
        }

    // ---- lin_kj * rbf_h ----
    zero8(acc);
    mm128(Aw, 136, WtKJ, acc);
#pragma unroll
    for (int t = 0; t < 8; ++t)
#pragma unroll
        for (int v = 0; v < 8; ++v) {
            int col = t * 16 + r;
            int mrow = wave * 16 + v + 8 * g;
            long gr = base + mrow;
            float val = silu_f(acc[t][v] + bkj_s[col]);
            float rh = 0.f;
#pragma unroll
            for (int q = 0; q < NRr; ++q) rh += rbf_s[mrow * NRr + q] * lrw_s[q * H128 + col];
            if (gr < E) x_kj[gr * H128 + col] = (__bf16)(val * rh);
        }
}

// ---------------- K2: triplet GEMM + scatter ----------------
// out_t = [sbf_h_j * xk_l]_{T x 1024} @ WtBig_{1024 x 128}; atomadd into agg[idx_ji]
__global__ __launch_bounds__(256) void k2_triplet(
    const float* __restrict__ sbf, const int* __restrict__ idx_kj,
    const int* __restrict__ idx_ji, const float* __restrict__ lin_sbf_w,
    const __bf16* __restrict__ WtBig, const __bf16* __restrict__ x_kj,
    float* __restrict__ agg, int T) {
    __shared__ __align__(16) __bf16 xkA[128 * 136];
    __shared__ __align__(16) __bf16 Bs[2][128 * 32];  // 8KB double-buffered B chunk
    __shared__ float sh[128 * NBb];
    __shared__ float lsw[SBFW * NBb];
    __shared__ int ikj[128], iji[128];

    const int tid = threadIdx.x;
    const long base = (long)blockIdx.x * 128;
    const int n_st = tid >> 1, part = tid & 1;  // B-chunk staging coords

    // prefetch B chunk 0 (async -> LDS), overlaps gather + sbf_h below
    {
        const __bf16* s = WtBig + (long)n_st * 1024 + part * 16;
        __bf16* d = &Bs[0][n_st * 32 + part * 16];
        async_ld16(lds_off32(d), s);
        async_ld16(lds_off32(d + 8), s + 8);
    }

    for (int i = tid; i < SBFW * NBb; i += 256) lsw[i] = lin_sbf_w[i];
    if (tid < 128) {
        long t0 = base + tid;
        ikj[tid] = (t0 < T) ? idx_kj[t0] : 0;
        iji[tid] = (t0 < T) ? idx_ji[t0] : -1;
    }
    __syncthreads();  // ikj ready for gather

    {   // gather xk rows (bf16), async global->LDS, 2 threads per row
        int row = tid >> 1, half = tid & 1;
        long t0 = base + row;
        __bf16* dst = xkA + row * 136 + half * 64;
        if (t0 < T) {
            const __bf16* src = x_kj + (long)ikj[row] * H128 + half * 64;
#pragma unroll
            for (int i = 0; i < 64; i += 8)
                async_ld16(lds_off32(dst + i), src + i);
        } else {
#pragma unroll
            for (int i = 0; i < 64; ++i) dst[i] = (__bf16)0.f;
        }
    }
    if (tid < 128) {  // sbf_h row per thread (overlaps with async gather)
        long t0 = base + tid;
        float a[NBb] = {0.f, 0.f, 0.f, 0.f, 0.f, 0.f, 0.f, 0.f};
        if (t0 < T) {
            const float* sp = sbf + t0 * SBFW;
#pragma unroll
            for (int q = 0; q < SBFW; ++q) {
                float s = sp[q];
#pragma unroll
                for (int j = 0; j < NBb; ++j) a[j] += s * lsw[q * NBb + j];
            }
        }
#pragma unroll
        for (int j = 0; j < NBb; ++j) sh[tid * NBb + j] = a[j];
    }

    const int wave = tid >> 5, lane = tid & 31, g = lane >> 4, r = lane & 15;
    const int mrow = wave * 16 + r;  // this lane's A row within the tile
    v8f acc[8];
    zero8(acc);

    for (int ks = 0; ks < 32; ++ks) {  // K = 1024 = 32 steps of 32
        async_wait0();     // own async chunk (+gather on first iter) landed
        __syncthreads();   // everyone's portion visible; prior readers retired
        if (ks + 1 < 32) {  // prefetch next chunk into other buffer
            const __bf16* s = WtBig + (long)n_st * 1024 + (ks + 1) * 32 + part * 16;
            __bf16* d = &Bs[(ks + 1) & 1][n_st * 32 + part * 16];
            async_ld16(lds_off32(d), s);
            async_ld16(lds_off32(d + 8), s + 8);
        }

        // A fragment: scale xk by sbf_h[row, j], j = ks/4
        float s = sh[mrow * NBb + (ks >> 2)];
        const __bf16* ar = xkA + mrow * 136 + (ks & 3) * 32;
        v8bf a0 = *(const v8bf*)(ar + 8 * g);
        v8bf a1 = *(const v8bf*)(ar + 16 + 8 * g);
        v16bf a;
#pragma unroll
        for (int i = 0; i < 8; ++i) {
            a[i]     = (__bf16)((float)a0[i] * s);
            a[i + 8] = (__bf16)((float)a1[i] * s);
        }
        const __bf16* Bcur = Bs[ks & 1];
#pragma unroll
        for (int t = 0; t < 8; ++t) {
            const __bf16* br = Bcur + (t * 16 + r) * 32 + 8 * g;
            v16bf b = cat8(*(const v8bf*)(br), *(const v8bf*)(br + 16));
            acc[t] = wmma_bf16(a, b, acc[t]);
        }
    }

    // scatter-add: agg[idx_ji[row], col] += acc
#pragma unroll
    for (int t = 0; t < 8; ++t)
#pragma unroll
        for (int v = 0; v < 8; ++v) {
            int M = wave * 16 + v + 8 * g;
            int e = iji[M];
            if (e >= 0) atomicAdd(agg + (long)e * H128 + t * 16 + r, acc[t][v]);
        }
}

// ---------------- K3: edge output chain ----------------
__global__ __launch_bounds__(256) void k3_edge_out(
    const float* __restrict__ x, const float* __restrict__ x_ji,
    const float* __restrict__ agg,
    const __bf16* __restrict__ Wrb1, const __bf16* __restrict__ Wrb2,
    const __bf16* __restrict__ Wlin, const __bf16* __restrict__ Wra11,
    const __bf16* __restrict__ Wra12, const __bf16* __restrict__ Wra21,
    const __bf16* __restrict__ Wra22, const __bf16* __restrict__ Wout,
    const float* __restrict__ brb1, const float* __restrict__ brb2,
    const float* __restrict__ blin, const float* __restrict__ bra11,
    const float* __restrict__ bra12, const float* __restrict__ bra21,
    const float* __restrict__ bra22, const float* __restrict__ bout,
    float* __restrict__ out, int E) {
    __shared__ __align__(16) __bf16 hA[8][16 * 136];  // per-wave transpose scratch

    const int tid = threadIdx.x;
    const int wave = tid >> 5, lane = tid & 31, g = lane >> 4, r = lane & 15;
    const long base = (long)blockIdx.x * 128;
    __bf16* Aw = hA[wave];

    v8f h[8], acc[8];

    // h0 = x_ji + agg (C layout)
#pragma unroll
    for (int t = 0; t < 8; ++t)
#pragma unroll
        for (int v = 0; v < 8; ++v) {
            int col = t * 16 + r;
            long gr = base + wave * 16 + v + 8 * g;
            long idx = gr * H128 + col;
            h[t][v] = (gr < E) ? (x_ji[idx] + agg[idx]) : 0.f;
        }

    // residual before skip
    linear_silu(Aw, h, Wrb1, brb1, acc);
    linear_silu(Aw, acc, Wrb2, brb2, acc);
#pragma unroll
    for (int t = 0; t < 8; ++t)
#pragma unroll
        for (int v = 0; v < 8; ++v) h[t][v] += acc[t][v];

    // h = silu(h @ lin + b) + x
    linear_silu(Aw, h, Wlin, blin, acc);
#pragma unroll
    for (int t = 0; t < 8; ++t)
#pragma unroll
        for (int v = 0; v < 8; ++v) {
            int col = t * 16 + r;
            long gr = base + wave * 16 + v + 8 * g;
            float xv = (gr < E) ? x[gr * H128 + col] : 0.f;
            h[t][v] = acc[t][v] + xv;
        }

    // residual after skip x2
    linear_silu(Aw, h, Wra11, bra11, acc);
    linear_silu(Aw, acc, Wra12, bra12, acc);
#pragma unroll
    for (int t = 0; t < 8; ++t)
#pragma unroll
        for (int v = 0; v < 8; ++v) h[t][v] += acc[t][v];

    linear_silu(Aw, h, Wra21, bra21, acc);
    linear_silu(Aw, acc, Wra22, bra22, acc);
#pragma unroll
    for (int t = 0; t < 8; ++t)
#pragma unroll
        for (int v = 0; v < 8; ++v) h[t][v] += acc[t][v];

    // out = silu(h @ lin_out + b)
    linear_silu(Aw, h, Wout, bout, acc);
#pragma unroll
    for (int t = 0; t < 8; ++t)
#pragma unroll
        for (int v = 0; v < 8; ++v) {
            int col = t * 16 + r;
            long gr = base + wave * 16 + v + 8 * g;
            if (gr < E) out[gr * H128 + col] = acc[t][v];
        }
}

// ---------------- host launch ----------------

extern "C" void kernel_launch(void* const* d_in, const int* in_sizes, int n_in,
                              void* d_out, int out_size, void* d_ws, size_t ws_size,
                              hipStream_t stream) {
    const int E = in_sizes[0] / H128;
    const int T = in_sizes[3];

    const float* x        = (const float*)d_in[0];
    const float* rbf      = (const float*)d_in[1];
    const float* sbf      = (const float*)d_in[2];
    const int*   idx_kj   = (const int*)d_in[3];
    const int*   idx_ji   = (const int*)d_in[4];
    const float* lin_rbf_w = (const float*)d_in[5];
    const float* lin_sbf_w = (const float*)d_in[6];
    const float* Wbig     = (const float*)d_in[11];

    char* ws = (char*)d_ws;
    __bf16* wtBig = (__bf16*)ws;                                   // 128*1024 bf16
    __bf16* mats  = (__bf16*)(ws + 262144);                        // 10 x 128*128 bf16
    size_t off = 262144 + 10 * 32768;
    float*  x_ji = (float*)(ws + off);                             // E*128 f32
    __bf16* x_kj = (__bf16*)(ws + off + (size_t)E * H128 * 4);     // E*128 bf16

    float* agg = (float*)d_out;  // reuse d_out as segment-sum accumulator

    // ---- prep: weights -> bf16 WMMA layouts ----
    cvt_bf16_kernel<<<512, 256, 0, stream>>>(Wbig, wtBig, H128 * NBb * H128);
    const int srcIdx[10] = {9, 7, 12, 14, 16, 18, 20, 22, 24, 26};
    for (int m = 0; m < 10; ++m)
        transpose_cvt_kernel<<<64, 256, 0, stream>>>((const float*)d_in[srcIdx[m]],
                                                     mats + m * 16384);
    // ---- zero agg ----
    zero_f32_kernel<<<2048, 256, 0, stream>>>(agg, (long)E * H128);

    // ---- K1 ----
    int gridE = (E + 127) / 128;
    k1_edge_in<<<gridE, 256, 0, stream>>>(x, rbf, lin_rbf_w,
                                          (const float*)d_in[8],   // lin_kj_b
                                          (const float*)d_in[10],  // lin_ji_b
                                          mats + 1 * 16384,        // WtKJ
                                          mats + 0 * 16384,        // WtJI
                                          x_ji, x_kj, E);

    // ---- K2 ----
    int gridT = (T + 127) / 128;
    k2_triplet<<<gridT, 256, 0, stream>>>(sbf, idx_kj, idx_ji, lin_sbf_w, wtBig,
                                          x_kj, agg, T);

    // ---- K3 ----
    k3_edge_out<<<gridE, 256, 0, stream>>>(
        x, x_ji, agg,
        mats + 2 * 16384, mats + 3 * 16384, mats + 4 * 16384, mats + 5 * 16384,
        mats + 6 * 16384, mats + 7 * 16384, mats + 8 * 16384, mats + 9 * 16384,
        (const float*)d_in[13], (const float*)d_in[15], (const float*)d_in[17],
        (const float*)d_in[19], (const float*)d_in[21], (const float*)d_in[23],
        (const float*)d_in[25], (const float*)d_in[27],
        (float*)d_out, E);

    (void)n_in; (void)out_size; (void)ws_size;
}
